// MultiHeadAttention_4337916970015
// MI455X (gfx1250) — matrix operand order repaired and checked by reading, not compiled, not run on hardware
//
#include <hip/hip_runtime.h>
#include <hip/hip_bf16.h>
#include <math.h>

// ---------------------------------------------------------------------------
// MultiHeadAttention forward for MI455X (gfx1250), wave32 + WMMA f16 + TDM.
// Shapes: bs=4, qlen=2048, dim=1024, heads=16, dh=64.
// ---------------------------------------------------------------------------

typedef __attribute__((ext_vector_type(16))) _Float16 v16h;
typedef __attribute__((ext_vector_type(8)))  _Float16 v8h;
typedef __attribute__((ext_vector_type(8)))  float    v8f;
typedef __attribute__((ext_vector_type(4)))  unsigned int u32x4;
typedef __attribute__((ext_vector_type(8)))  int      i32x8;
typedef __attribute__((ext_vector_type(4)))  int      i32x4;

#define BS    4
#define SEQ   2048
#define DIM   1024
#define NH    16
#define DH    64
#define MROWS (BS * SEQ)   // 8192

#if __has_builtin(__builtin_amdgcn_tensor_load_to_lds) && \
    __has_builtin(__builtin_amdgcn_s_wait_tensorcnt)
#define HAVE_TDM 1
#endif

union F16Frag {
    v16h     v;
    v8h      h2[2];
    _Float16 h[16];
};

// D = A(16x32 f16) x B(32x16 f16) + C(16x16 f32)
static __device__ __forceinline__ v8f wmma_f16(const F16Frag& a, const F16Frag& b, v8f c) {
    return __builtin_amdgcn_wmma_f32_16x16x32_f16(
        /*neg_a=*/false, a.v, /*neg_b=*/false, b.v,
        /*c_mod=*/(short)0, c, /*reuse_a=*/false, /*reuse_b=*/false);
}

#if defined(HAVE_TDM)
// Issue a 2-D TDM tile load (f16 elements): tile [tile1 rows x tile0 elems]
// from global (row stride stride0 elems) into LDS at byte offset lds_off.
// D# packing per CDNA5 ISA ch. 8.3/8.4 (groups 2/3 zero => 2-D tensor).
static __device__ __forceinline__ void tdm_load_2d_f16(
        unsigned int lds_off, const void* gptr,
        unsigned int dim0, unsigned int dim1,
        unsigned int tile0, unsigned int tile1,
        unsigned int stride0) {
    unsigned long long ga = (unsigned long long)gptr;
    u32x4 g0;
    g0[0] = 1u;                                            // count=1 (valid user D#)
    g0[1] = lds_off;                                       // lds_addr (bytes)
    g0[2] = (unsigned int)(ga & 0xffffffffull);            // global_addr[31:0]
    g0[3] = (unsigned int)((ga >> 32) & 0x1ffffffull)      // global_addr[56:32]
          | (2u << 30);                                    // type=2 ("image")
    i32x8 g1;
    g1[0] = (int)(1u << 16);                               // data_size=1 (2 bytes)
    g1[1] = (int)((dim0 & 0xffffu) << 16);                 // tensor_dim0[15:0]
    g1[2] = (int)(((dim0 >> 16) & 0xffffu) | ((dim1 & 0xffffu) << 16));
    g1[3] = (int)(((dim1 >> 16) & 0xffffu) | ((tile0 & 0xffffu) << 16));
    g1[4] = (int)(tile1 & 0xffffu);                        // tile_dim1 (tile_dim2=0)
    g1[5] = (int)stride0;                                  // tensor_dim0_stride[31:0]
    g1[6] = 0;                                             // stride0 hi / dim1_stride lo
    g1[7] = 0;
    i32x4 z4 = {0, 0, 0, 0};
#if __has_include(<hip/amd_detail/amd_gfx1250_TDM.h>) || (__clang_major__ >= 23)
    i32x8 z8 = {0, 0, 0, 0, 0, 0, 0, 0};
    __builtin_amdgcn_tensor_load_to_lds(g0, g1, z4, z4, z8, 0);
#else
    __builtin_amdgcn_tensor_load_to_lds(g0, g1, z4, z4, 0);
#endif
}
static __device__ __forceinline__ unsigned int lds_offset_of(const void* p) {
    return (unsigned int)(unsigned long long)p;  // flat LDS addr[31:0] == LDS offset
}
#endif

// ---------------------------------------------------------------------------
// f32 -> f16 conversion (grid-stride)
// ---------------------------------------------------------------------------
__global__ void mha_cvt_f16_kernel(const float* __restrict__ in,
                                   _Float16* __restrict__ out, size_t n) {
    size_t i = (size_t)blockIdx.x * blockDim.x + threadIdx.x;
    size_t stride = (size_t)gridDim.x * blockDim.x;
    for (; i < n; i += stride) out[i] = (_Float16)in[i];
}

// ---------------------------------------------------------------------------
// WMMA GEMM:  Out = X(MxK f16, row-major) @ W(NxK f16, row-major)^T + bias
// Block tile 64(M) x 64(N); 128 threads = 4 waves; wave w owns 16 M-rows and
// all 4 N-subtiles. B-panel (64x32 per K-step) staged through LDS with a
// double-buffered Tensor-Data-Mover pipeline (wave 0 issues, s_wait_tensorcnt
// + barrier closes each stage); A streams from global with prefetch.
// MODE 0: Q -> scale 1/8, store f16 [b,h,s,dh]
// MODE 1: K ->            store f16 [b,h,s,dh]
// MODE 2: V ->            store f16 [b,h,dh,s]  (transposed for PV B-operand)
// MODE 3: O ->            store f32 row-major to d_out
// ---------------------------------------------------------------------------
template <int MODE>
__global__ __launch_bounds__(128)
void mha_gemm_wmma_kernel(const _Float16* __restrict__ X,
                          const _Float16* __restrict__ W,
                          const float* __restrict__ bias,
                          void* __restrict__ outp) {
    const int lane = threadIdx.x & 31;
    const int wave = threadIdx.x >> 5;
    const int m0 = blockIdx.x * 64 + wave * 16;
    const int n0 = blockIdx.y * 64;

    // A fragment: lane (m = lane&15) loads 16 consecutive K [kb..kb+15],
    // kb = (lane>>4)*16  (matches B's contiguous K order).
    const _Float16* arow = X + (size_t)(m0 + (lane & 15)) * DIM + ((lane >> 4) * 16);

    v8f acc[4] = {};

#if defined(HAVE_TDM)
    __shared__ _Float16 Bt[2][64 * 32];   // two 4KB B-panels [64 rows][32 k]
    const _Float16* wtile = W + (size_t)n0 * DIM;
    if (wave == 0) {
        tdm_load_2d_f16(lds_offset_of(&Bt[0][0]), wtile, 32, 64, 32, 64, DIM);
        __builtin_amdgcn_s_wait_tensorcnt(0);
    }
    __syncthreads();

    for (int it = 0; it < DIM / 32; ++it) {
        const int k0 = it * 32;
        const int cur = it & 1;
        if (wave == 0 && (it + 1) < DIM / 32)
            tdm_load_2d_f16(lds_offset_of(&Bt[cur ^ 1][0]),
                            wtile + (k0 + 32), 32, 64, 32, 64, DIM);

        __builtin_prefetch(arow + k0 + 256, 0, 0);   // global_prefetch_b8
        F16Frag a;
        a.h2[0] = *(const v8h*)(arow + k0);
        a.h2[1] = *(const v8h*)(arow + k0 + 8);
        // B fragment from LDS: lane holds column n = lane&15 of subtile,
        // 16 consecutive K at (lane>>4)*16; row pitch 32 halves.
        const _Float16* bl = &Bt[cur][0] + (size_t)(lane & 15) * 32 + ((lane >> 4) * 16);
#pragma unroll
        for (int nt = 0; nt < 4; ++nt) {
            F16Frag b;
            b.v = *(const v16h*)(bl + nt * 16 * 32);
            acc[nt] = wmma_f16(a, b, acc[nt]);
        }
        if (wave == 0) __builtin_amdgcn_s_wait_tensorcnt(0);
        __syncthreads();
    }
#else
    const _Float16* brow[4];
#pragma unroll
    for (int nt = 0; nt < 4; ++nt)
        brow[nt] = W + (size_t)(n0 + nt * 16 + (lane & 15)) * DIM + ((lane >> 4) * 16);
    for (int k0 = 0; k0 < DIM; k0 += 32) {
        __builtin_prefetch(arow + k0 + 256, 0, 0);
        __builtin_prefetch(brow[0] + k0 + 256, 0, 0);
        F16Frag a;
        a.h2[0] = *(const v8h*)(arow + k0);
        a.h2[1] = *(const v8h*)(arow + k0 + 8);
#pragma unroll
        for (int nt = 0; nt < 4; ++nt) {
            F16Frag b;
            b.v = *(const v16h*)(brow[nt] + k0);
            acc[nt] = wmma_f16(a, b, acc[nt]);
        }
    }
#endif

    // C/D layout: lane<16 -> N=lane, M=r; lane>=16 -> N=lane-16, M=r+8.
    const int r0  = (lane >> 4) * 8;
    const int col = lane & 15;
#pragma unroll
    for (int nt = 0; nt < 4; ++nt) {
#pragma unroll
        for (int r = 0; r < 8; ++r) {
            const int gm = m0 + r0 + r;        // row in [0, MROWS)
            const int gn = n0 + nt * 16 + col; // col in [0, DIM)
            float val = acc[nt][r] + bias[gn];
            const int b = gm >> 11, s = gm & (SEQ - 1);
            const int h = gn >> 6,  d = gn & (DH - 1);
            if constexpr (MODE == 0) {
                val *= 0.125f; // 1/sqrt(dh)
                ((_Float16*)outp)[(((size_t)b * NH + h) * SEQ + s) * DH + d] = (_Float16)val;
            } else if constexpr (MODE == 1) {
                ((_Float16*)outp)[(((size_t)b * NH + h) * SEQ + s) * DH + d] = (_Float16)val;
            } else if constexpr (MODE == 2) {
                ((_Float16*)outp)[(((size_t)b * NH + h) * DH + d) * SEQ + s] = (_Float16)val;
            } else {
                ((float*)outp)[(size_t)gm * DIM + gn] = val;
            }
        }
    }
}

// ---------------------------------------------------------------------------
// Attention core: one workgroup = (b, h, 16 query rows).
// Phase 0: TDM-load the 16x64 q-tile into LDS.
// Phase 1: scores S[16][2048] = q kT via WMMA (dh=64 -> 2 K-steps) into LDS.
// Phase 2: masked softmax per row (wave handles 2 rows), f16 weights -> LDS,
//          head-averaged weights accumulated to accW via f32 atomics.
// Phase 3: context[16][64] = W @ V via WMMA (K=2048), V pre-transposed.
// Dynamic LDS: 16*2048*4 (scores) + 16*2048*2 (weights) + 2KB (q tile).
// ---------------------------------------------------------------------------
__global__ __launch_bounds__(256)
void mha_attn_kernel(const _Float16* __restrict__ qbuf,
                     const _Float16* __restrict__ kbuf,
                     const _Float16* __restrict__ vT,
                     const int* __restrict__ mask,
                     _Float16* __restrict__ ctx,
                     float* __restrict__ accW) {
    extern __shared__ char smem[];
    float*    S  = (float*)smem;                                  // [16][SEQ]
    _Float16* WH = (_Float16*)(smem + 16 * SEQ * sizeof(float));  // [16][SEQ]
    _Float16* QT = (_Float16*)(smem + 16 * SEQ * sizeof(float)
                                    + 16 * SEQ * sizeof(_Float16)); // [16][DH]

    const int lane = threadIdx.x & 31;
    const int wave = threadIdx.x >> 5;          // 0..7
    const int qt = blockIdx.x;                  // query tile (16 rows)
    const int h  = blockIdx.y;
    const int b  = blockIdx.z;

    const _Float16* qbase = qbuf + (((size_t)b * NH + h) * SEQ + qt * 16) * DH;
    const _Float16* kbase = kbuf + (((size_t)b * NH + h) * SEQ) * DH;

    // ---- Phase 1: scores -------------------------------------------------
    {
#if defined(HAVE_TDM)
        if (wave == 0) {
            tdm_load_2d_f16(lds_offset_of(QT), qbase, DH, 16, DH, 16, DH);
            __builtin_amdgcn_s_wait_tensorcnt(0);
        }
        __syncthreads();
        const _Float16* ap = QT + (size_t)(lane & 15) * DH + ((lane >> 4) * 16);
#else
        const _Float16* ap = qbase + (size_t)(lane & 15) * DH + ((lane >> 4) * 16);
#endif
        F16Frag a0, a1;
        a0.h2[0] = *(const v8h*)(ap);       a0.h2[1] = *(const v8h*)(ap + 8);
        a1.h2[0] = *(const v8h*)(ap + 32);  a1.h2[1] = *(const v8h*)(ap + 40);

        const int r0 = (lane >> 4) * 8, col = lane & 15;
        for (int jt = wave; jt < SEQ / 16; jt += 8) {   // 128 key tiles / 8 waves
            const _Float16* bp = kbase + (size_t)(jt * 16 + (lane & 15)) * DH + ((lane >> 4) * 16);
            F16Frag b0, b1;
            b0.v = *(const v16h*)(bp);
            b1.v = *(const v16h*)(bp + 32);
            v8f c = {};
            c = wmma_f16(a0, b0, c);
            c = wmma_f16(a1, b1, c);
#pragma unroll
            for (int r = 0; r < 8; ++r)
                S[(size_t)(r0 + r) * SEQ + jt * 16 + col] = c[r];
        }
    }
    __syncthreads();

    // ---- Phase 2: softmax + weight accumulation --------------------------
    const int* mrow = mask + (size_t)b * SEQ;
#pragma unroll
    for (int rr = 0; rr < 2; ++rr) {
        const int r = wave * 2 + rr;
        const int grow = qt * 16 + r;
        float* Srow = S + (size_t)r * SEQ;

        float m = -3.402823466e+38f;
        for (int c = lane; c < SEQ; c += 32)
            if (mrow[c] != 0) m = fmaxf(m, Srow[c]);
        for (int off = 16; off > 0; off >>= 1) m = fmaxf(m, __shfl_xor(m, off, 32));

        float sum = 0.f;
        for (int c = lane; c < SEQ; c += 32) {
            float e = (mrow[c] != 0) ? __expf(Srow[c] - m) : 0.f;
            Srow[c] = e;
            sum += e;
        }
        for (int off = 16; off > 0; off >>= 1) sum += __shfl_xor(sum, off, 32);

        const float inv = 1.f / sum;
        float* arow = accW + ((size_t)b * SEQ + grow) * SEQ;
        for (int c = lane; c < SEQ; c += 32) {
            float w = Srow[c] * inv;
            WH[(size_t)r * SEQ + c] = (_Float16)w;
            __hip_atomic_fetch_add(&arow[c], w * (1.f / NH),
                                   __ATOMIC_RELAXED, __HIP_MEMORY_SCOPE_AGENT);
        }
    }
    __syncthreads();

    // ---- Phase 3: context = W @ V  (waves 0..3, one 16-col d-tile each) ---
    if (wave < 4) {
        const int d0 = wave * 16;
        const _Float16* arow = WH + (size_t)(lane & 15) * SEQ + ((lane >> 4) * 16);
        const _Float16* vrow = vT + (((size_t)b * NH + h) * DH + d0 + (lane & 15)) * SEQ
                                  + ((lane >> 4) * 16);
        v8f c = {};
        for (int k0 = 0; k0 < SEQ; k0 += 32) {
            F16Frag a, bb;
            a.h2[0] = *(const v8h*)(arow + k0);
            a.h2[1] = *(const v8h*)(arow + k0 + 8);
            bb.v    = *(const v16h*)(vrow + k0);
            c = wmma_f16(a, bb, c);
        }
        const int r0 = (lane >> 4) * 8, col = lane & 15;
#pragma unroll
        for (int r = 0; r < 8; ++r) {
            const int row = qt * 16 + r0 + r;
            ctx[((size_t)b * SEQ + row) * DIM + h * DH + d0 + col] = (_Float16)c[r];
        }
    }
}

// ---------------------------------------------------------------------------
// Argmax over k of head-summed weights; first-index tie-break (jnp.argmax).
// One wave per (b, q) row.
// ---------------------------------------------------------------------------
__global__ __launch_bounds__(256)
void mha_argmax_kernel(const float* __restrict__ accW, float* __restrict__ outIdx) {
    const int lane = threadIdx.x & 31;
    const int wave = threadIdx.x >> 5;
    const int row = blockIdx.x * 8 + wave;        // [0, BS*SEQ)
    const float* p = accW + (size_t)row * SEQ;
    float best = -3.402823466e+38f;
    int   bi   = 0x7fffffff;
    for (int c = lane; c < SEQ; c += 32) {
        float v = p[c];
        if (v > best || (v == best && c < bi)) { best = v; bi = c; }
    }
    for (int off = 16; off > 0; off >>= 1) {
        float ov = __shfl_xor(best, off, 32);
        int   oi = __shfl_xor(bi, off, 32);
        if (ov > best || (ov == best && oi < bi)) { best = ov; bi = oi; }
    }
    if (lane == 0) outIdx[row] = (float)bi;
}

// ---------------------------------------------------------------------------
extern "C" void kernel_launch(void* const* d_in, const int* in_sizes, int n_in,
                              void* d_out, int out_size, void* d_ws, size_t ws_size,
                              hipStream_t stream) {
    const float* query = (const float*)d_in[0];
    const int*   mask  = (const int*)d_in[1];
    const float* Wq = (const float*)d_in[2]; const float* bq = (const float*)d_in[3];
    const float* Wk = (const float*)d_in[4]; const float* bk = (const float*)d_in[5];
    const float* Wv = (const float*)d_in[6]; const float* bv = (const float*)d_in[7];
    const float* Wo = (const float*)d_in[8]; const float* bo = (const float*)d_in[9];

    size_t off = 0;
    auto take = [&](size_t bytes) -> void* {
        void* p = (char*)d_ws + off;
        off += (bytes + 255) & ~(size_t)255;
        return p;
    };
    _Float16* X16  = (_Float16*)take((size_t)MROWS * DIM * 2); // 16 MB
    _Float16* Wq16 = (_Float16*)take((size_t)DIM * DIM * 2);   // 2 MB
    _Float16* Wk16 = (_Float16*)take((size_t)DIM * DIM * 2);
    _Float16* Wv16 = (_Float16*)take((size_t)DIM * DIM * 2);
    _Float16* Wo16 = (_Float16*)take((size_t)DIM * DIM * 2);
    _Float16* qb   = (_Float16*)take((size_t)MROWS * DIM * 2); // [b,h,s,dh]
    _Float16* kb   = (_Float16*)take((size_t)MROWS * DIM * 2); // [b,h,s,dh]
    _Float16* vT   = (_Float16*)take((size_t)MROWS * DIM * 2); // [b,h,dh,s]
    _Float16* ctx  = (_Float16*)take((size_t)MROWS * DIM * 2); // [b,s,dim]
    float*    accW = (float*)take((size_t)BS * SEQ * SEQ * 4); // 64 MB

    hipMemsetAsync(accW, 0, (size_t)BS * SEQ * SEQ * 4, stream);

    mha_cvt_f16_kernel<<<512, 256, 0, stream>>>(query, X16, (size_t)MROWS * DIM);
    mha_cvt_f16_kernel<<<128, 256, 0, stream>>>(Wq, Wq16, (size_t)DIM * DIM);
    mha_cvt_f16_kernel<<<128, 256, 0, stream>>>(Wk, Wk16, (size_t)DIM * DIM);
    mha_cvt_f16_kernel<<<128, 256, 0, stream>>>(Wv, Wv16, (size_t)DIM * DIM);
    mha_cvt_f16_kernel<<<128, 256, 0, stream>>>(Wo, Wo16, (size_t)DIM * DIM);

    dim3 ggrid(MROWS / 64, DIM / 64);
    mha_gemm_wmma_kernel<0><<<ggrid, 128, 0, stream>>>(X16, Wq16, bq, qb);
    mha_gemm_wmma_kernel<1><<<ggrid, 128, 0, stream>>>(X16, Wk16, bk, kb);
    mha_gemm_wmma_kernel<2><<<ggrid, 128, 0, stream>>>(X16, Wv16, bv, vT);

    const size_t lds = (size_t)16 * SEQ * sizeof(float)
                     + (size_t)16 * SEQ * sizeof(_Float16)
                     + (size_t)16 * DH * sizeof(_Float16);
    mha_attn_kernel<<<dim3(SEQ / 16, NH, BS), 256, lds, stream>>>(qb, kb, vT, mask, ctx, accW);

    mha_gemm_wmma_kernel<3><<<ggrid, 128, 0, stream>>>(ctx, Wo16, bo, d_out);

    mha_argmax_kernel<<<MROWS / 8, 256, 0, stream>>>(accW, (float*)d_out + (size_t)MROWS * DIM);
}
